// VAE_Spike_GLM_81140522156035
// MI455X (gfx1250) — matrix operand order repaired and checked by reading, not compiled, not run on hardware
//
#include <hip/hip_runtime.h>
#include <math.h>
#include <stdint.h>

// ---------------------------------------------------------------------------
// Problem constants (from reference setup_inputs)
// ---------------------------------------------------------------------------
#define TDATA 30000
#define ENO   1000
#define INO   250
#define SUB   20
#define TNO   200
#define NB    13
#define PI_F  3.14159265358979323846f

typedef __attribute__((ext_vector_type(16))) _Float16 v16h;
typedef __attribute__((ext_vector_type(8)))  _Float16 v8h;
typedef __attribute__((ext_vector_type(8)))  float    v8f;
typedef __attribute__((ext_vector_type(4)))  unsigned v4u;
typedef __attribute__((ext_vector_type(8)))  int      v8i;
typedef __attribute__((ext_vector_type(4)))  int      v4i;

// ---- Tensor Data Mover availability / arity detection ----------------------
#if defined(__HIP_DEVICE_COMPILE__) && defined(__has_builtin)
#if __has_builtin(__builtin_amdgcn_tensor_load_to_lds)
#define HAVE_TDM 1
#endif
#endif
#ifndef HAVE_TDM
#define HAVE_TDM 0
#endif
#if __has_include(<hip/amd_detail/amd_gfx1250_TDM.h>)
#define TDM_SIX_ARGS 1
#else
#define TDM_SIX_ARGS 0
#endif

#if HAVE_TDM
// 1-D contiguous global->LDS DMA: elems f32 words starting at gsrc -> lds_dst.
__device__ __forceinline__ void tdm_load_1d_f32(const float* gsrc,
                                                float* lds_dst, int elems) {
  typedef __attribute__((address_space(3))) float lds_f;
  unsigned lds_off = (unsigned)(uintptr_t)(lds_f*)lds_dst;
  unsigned long long ga = (unsigned long long)(uintptr_t)gsrc;
  v4u g0;
  g0[0] = 1u;                                   // count=1 (valid user D#)
  g0[1] = lds_off;                              // lds_addr (bytes)
  g0[2] = (unsigned)(ga & 0xFFFFFFFFu);         // global_addr[31:0]
  g0[3] = (unsigned)((ga >> 32) & 0x01FFFFFFull) | 0x80000000u;  // type=2
  v8i g1;
  g1[0] = 0x00020000;                           // data_size=4B, no multicast
  g1[1] = (int)(((unsigned)elems & 0xFFFFu) << 16);   // tensor_dim0 lo
  g1[2] = (int)(((unsigned)elems >> 16) | (1u << 16)); // tensor_dim0 hi | dim1=1
  g1[3] = (int)(((unsigned)elems & 0xFFFFu) << 16);   // tile_dim0
  g1[4] = 0;                                    // tile_dim1=0, tile_dim2=0
  g1[5] = elems;                                // tensor_dim0_stride lo
  g1[6] = 0;
  g1[7] = 0;
  v4i z4 = {0, 0, 0, 0};
#if TDM_SIX_ARGS
  v8i z8 = {0, 0, 0, 0, 0, 0, 0, 0};
  __builtin_amdgcn_tensor_load_to_lds(g0, g1, z4, z4, z8, 0);
#else
  __builtin_amdgcn_tensor_load_to_lds(g0, g1, z4, z4, 0);
#endif
}
#endif  // HAVE_TDM

// ---------------------------------------------------------------------------
// Kernel 0: precompute cosine-basis kernels, C_den child lists, and the
// zero-padded f16 copies of C_syn_e / C_syn_i used as WMMA B-operands.
// ---------------------------------------------------------------------------
__global__ __launch_bounds__(256) void k_precompute(
    const float* __restrict__ Wsyn, const float* __restrict__ Whist,
    const float* __restrict__ Wprop, const float* __restrict__ Cden,
    const float* __restrict__ Ce, const float* __restrict__ Ci,
    float* __restrict__ EK, float* __restrict__ IK,
    float* __restrict__ HK, float* __restrict__ PK,
    _Float16* __restrict__ CeH, _Float16* __restrict__ CiH,
    int* __restrict__ ccnt, int* __restrict__ cidx, float* __restrict__ cw)
{
  int tid = threadIdx.x;
  for (int idx = tid; idx < SUB * TNO; idx += 256) {
    int s = idx / TNO;
    int tau = idx % TNO;
    float raw = 5.0f * logf((float)tau + 1.0f + 1e-8f);
    float e = 0.f, ii = 0.f, h = 0.f, p = 0.f;
    #pragma unroll
    for (int b = 0; b < NB; ++b) {
      float phi = 0.5f * PI_F * (float)b;
      float bv = 0.f;
      if (raw >= phi - PI_F && raw <= phi + PI_F)
        bv = 0.5f * cosf(raw - phi) + 0.5f;
      e  += Wsyn[(s * NB + b) * 2 + 0] * bv;
      ii += Wsyn[(s * NB + b) * 2 + 1] * bv;
      h  += Whist[s * NB + b] * bv;
      p  += Wprop[s * NB + b] * bv;
    }
    EK[idx] = e; IK[idx] = ii; HK[idx] = h; PK[idx] = p;
  }
  // Padded f16 B-operands: 32 x 1024 and 32 x 256, zeros outside (20 x K).
  for (int idx = tid; idx < 32 * 1024; idx += 256) {
    int r = idx >> 10, c = idx & 1023;
    CeH[idx] = (r < SUB && c < ENO) ? (_Float16)Ce[r * ENO + c] : (_Float16)0.f;
  }
  for (int idx = tid; idx < 32 * 256; idx += 256) {
    int r = idx >> 8, c = idx & 255;
    CiH[idx] = (r < SUB && c < INO) ? (_Float16)Ci[r * INO + c] : (_Float16)0.f;
  }
  // Children of target s: columns c with C_den[s,c] != 0 (<=8 kept)
  if (tid < SUB) {
    int cnt = 0;
    for (int c = 0; c < SUB; ++c) {
      float w = Cden[tid * SUB + c];
      if (w != 0.f && cnt < 8) {
        cidx[tid * 8 + cnt] = c;
        cw[tid * 8 + cnt] = w;
        ++cnt;
      }
    }
    ccnt[tid] = cnt;
  }
}

// ---------------------------------------------------------------------------
// Kernel 1: aggregation GEMMs via WMMA f16 (exact: inputs are 0/1, sums <= 50)
//   aggE = S_e @ C_syn_e.T   (30000 x 20),  aggI = S_i @ C_syn_i.T
// One wave per 16-row M-tile; two 16-col N-tiles cover SUB=20.
// Main K-loops are branchless (vector f32 loads + padded-f16 B); only the
// K-tail chunk of each GEMM needs special handling.
// ---------------------------------------------------------------------------
__global__ __launch_bounds__(256) void k_agg_wmma(
    const float* __restrict__ Se, const float* __restrict__ Si,
    const _Float16* __restrict__ CeH, const _Float16* __restrict__ CiH,
    float* __restrict__ aggE, float* __restrict__ aggI)
{
  const int lane = threadIdx.x & 31;
  const int wave = threadIdx.x >> 5;
  const int tile = blockIdx.x * 8 + wave;
  if (tile >= TDATA / 16) return;          // no barriers in this kernel
  const int t0 = tile * 16;

  // A layout (16x32 f16): lane -> row m = lane&15; kb = 0 / 8 for half-waves;
  // v16h idx i: K = kb+i (i<8), K = kb+8+i (i>=8)  => runs [kb,kb+8) [kb+16,kb+24)
  const int m  = lane & 15;
  const int kb = (lane < 16) ? 0 : 8;
  // B layout (32x16 f16): lane -> col n = lane&15; kh = 0 / 8; K = kh+i (i<8), kh+8+i (i>=8).
  const int nn = lane & 15;
  const int kh = (lane < 16) ? 0 : 8;

  v8f e0 = {}, e1 = {}, i0 = {}, i1 = {};

  // ---------------- excitatory GEMM: K = 1000 (31 full chunks + tail 8) ----
  {
    const float4* Arow4 = (const float4*)(Se + (size_t)(t0 + m) * ENO);
    const _Float16* Br0 = CeH + nn * 1024 + kh;
    const _Float16* Br1 = CeH + (16 + nn) * 1024 + kh;
    for (int k0 = 0; k0 < 992; k0 += 32) {
      int q = (k0 + kb) >> 2;
      float4 r0 = Arow4[q],     r1 = Arow4[q + 1];
      float4 r2 = Arow4[q + 4], r3 = Arow4[q + 5];
      v16h a;
      a[0]=(_Float16)r0.x;  a[1]=(_Float16)r0.y;  a[2]=(_Float16)r0.z;  a[3]=(_Float16)r0.w;
      a[4]=(_Float16)r1.x;  a[5]=(_Float16)r1.y;  a[6]=(_Float16)r1.z;  a[7]=(_Float16)r1.w;
      a[8]=(_Float16)r2.x;  a[9]=(_Float16)r2.y;  a[10]=(_Float16)r2.z; a[11]=(_Float16)r2.w;
      a[12]=(_Float16)r3.x; a[13]=(_Float16)r3.y; a[14]=(_Float16)r3.z; a[15]=(_Float16)r3.w;
      v16h b0 = __builtin_shufflevector(*(const v8h*)(Br0 + k0), *(const v8h*)(Br0 + k0 + 16), 0,1,2,3,4,5,6,7,8,9,10,11,12,13,14,15);
      v16h b1 = __builtin_shufflevector(*(const v8h*)(Br1 + k0), *(const v8h*)(Br1 + k0 + 16), 0,1,2,3,4,5,6,7,8,9,10,11,12,13,14,15);
      e0 = __builtin_amdgcn_wmma_f32_16x16x32_f16(false, a, false, b0,
                                                  (short)0, e0, false, false);
      e1 = __builtin_amdgcn_wmma_f32_16x16x32_f16(false, a, false, b1,
                                                  (short)0, e1, false, false);
    }
    {  // tail chunk k0 = 992: only lanes with kb==0 have valid K (992..999)
      v16h a = {};
      if (kb == 0) {
        float4 r0 = Arow4[248], r1 = Arow4[249];
        a[0]=(_Float16)r0.x; a[1]=(_Float16)r0.y; a[2]=(_Float16)r0.z; a[3]=(_Float16)r0.w;
        a[4]=(_Float16)r1.x; a[5]=(_Float16)r1.y; a[6]=(_Float16)r1.z; a[7]=(_Float16)r1.w;
      }
      v16h b0 = __builtin_shufflevector(*(const v8h*)(Br0 + 992), *(const v8h*)(Br0 + 1008), 0,1,2,3,4,5,6,7,8,9,10,11,12,13,14,15);
      v16h b1 = __builtin_shufflevector(*(const v8h*)(Br1 + 992), *(const v8h*)(Br1 + 1008), 0,1,2,3,4,5,6,7,8,9,10,11,12,13,14,15);
      e0 = __builtin_amdgcn_wmma_f32_16x16x32_f16(false, a, false, b0,
                                                  (short)0, e0, false, false);
      e1 = __builtin_amdgcn_wmma_f32_16x16x32_f16(false, a, false, b1,
                                                  (short)0, e1, false, false);
    }
  }
  // ---------------- inhibitory GEMM: K = 250 (7 full chunks + tail 26) -----
  {
    const float2* Arow2 = (const float2*)(Si + (size_t)(t0 + m) * INO);
    const _Float16* Br0 = CiH + nn * 256 + kh;
    const _Float16* Br1 = CiH + (16 + nn) * 256 + kh;
    for (int k0 = 0; k0 < 224; k0 += 32) {
      int q = (k0 + kb) >> 1;
      float2 r0 = Arow2[q],     r1 = Arow2[q + 1];
      float2 r2 = Arow2[q + 2], r3 = Arow2[q + 3];
      float2 r4 = Arow2[q + 8], r5 = Arow2[q + 9];
      float2 r6 = Arow2[q + 10], r7 = Arow2[q + 11];
      v16h a;
      a[0]=(_Float16)r0.x;  a[1]=(_Float16)r0.y;  a[2]=(_Float16)r1.x;  a[3]=(_Float16)r1.y;
      a[4]=(_Float16)r2.x;  a[5]=(_Float16)r2.y;  a[6]=(_Float16)r3.x;  a[7]=(_Float16)r3.y;
      a[8]=(_Float16)r4.x;  a[9]=(_Float16)r4.y;  a[10]=(_Float16)r5.x; a[11]=(_Float16)r5.y;
      a[12]=(_Float16)r6.x; a[13]=(_Float16)r6.y; a[14]=(_Float16)r7.x; a[15]=(_Float16)r7.y;
      v16h b0 = __builtin_shufflevector(*(const v8h*)(Br0 + k0), *(const v8h*)(Br0 + k0 + 16), 0,1,2,3,4,5,6,7,8,9,10,11,12,13,14,15);
      v16h b1 = __builtin_shufflevector(*(const v8h*)(Br1 + k0), *(const v8h*)(Br1 + k0 + 16), 0,1,2,3,4,5,6,7,8,9,10,11,12,13,14,15);
      i0 = __builtin_amdgcn_wmma_f32_16x16x32_f16(false, a, false, b0,
                                                  (short)0, i0, false, false);
      i1 = __builtin_amdgcn_wmma_f32_16x16x32_f16(false, a, false, b1,
                                                  (short)0, i1, false, false);
    }
    {  // tail chunk k0 = 224: run1 (224+kb..+7) fully valid; run2 partial
      v16h a = {};
      int q = (224 + kb) >> 1;
      float2 r0 = Arow2[q], r1 = Arow2[q + 1], r2 = Arow2[q + 2], r3 = Arow2[q + 3];
      a[0]=(_Float16)r0.x; a[1]=(_Float16)r0.y; a[2]=(_Float16)r1.x; a[3]=(_Float16)r1.y;
      a[4]=(_Float16)r2.x; a[5]=(_Float16)r2.y; a[6]=(_Float16)r3.x; a[7]=(_Float16)r3.y;
      if (kb == 0) {  // K 240..247 fully valid
        float2 r4 = Arow2[120], r5 = Arow2[121], r6 = Arow2[122], r7 = Arow2[123];
        a[8]=(_Float16)r4.x;  a[9]=(_Float16)r4.y;  a[10]=(_Float16)r5.x; a[11]=(_Float16)r5.y;
        a[12]=(_Float16)r6.x; a[13]=(_Float16)r6.y; a[14]=(_Float16)r7.x; a[15]=(_Float16)r7.y;
      } else {        // K 248,249 only
        float2 r4 = Arow2[124];
        a[8]=(_Float16)r4.x; a[9]=(_Float16)r4.y;
      }
      v16h b0 = __builtin_shufflevector(*(const v8h*)(Br0 + 224), *(const v8h*)(Br0 + 240), 0,1,2,3,4,5,6,7,8,9,10,11,12,13,14,15);
      v16h b1 = __builtin_shufflevector(*(const v8h*)(Br1 + 224), *(const v8h*)(Br1 + 240), 0,1,2,3,4,5,6,7,8,9,10,11,12,13,14,15);
      i0 = __builtin_amdgcn_wmma_f32_16x16x32_f16(false, a, false, b0,
                                                  (short)0, i0, false, false);
      i1 = __builtin_amdgcn_wmma_f32_16x16x32_f16(false, a, false, b1,
                                                  (short)0, i1, false, false);
    }
  }

  // C/D layout (16x16 f32): lane -> col lane&15; VGPR r -> row r + (lane<16?0:8)
  const int mb = (lane < 16) ? 0 : 8;
  const int n1 = 16 + nn;
  #pragma unroll
  for (int r = 0; r < 8; ++r) {
    int row = t0 + mb + r;
    aggE[(size_t)row * SUB + nn] = e0[r];
    aggI[(size_t)row * SUB + nn] = i0[r];
    if (n1 < SUB) {
      aggE[(size_t)row * SUB + n1] = e1[r];
      aggI[(size_t)row * SUB + n1] = i1[r];
    }
  }
}

// ---------------------------------------------------------------------------
// Kernel 2: causal 200-tap conv per subunit + Theta fold
// ---------------------------------------------------------------------------
__global__ __launch_bounds__(256) void k_conv(
    const float* __restrict__ aggE, const float* __restrict__ aggI,
    const float* __restrict__ EK, const float* __restrict__ IK,
    const float* __restrict__ Theta, float* __restrict__ SYN)
{
  __shared__ float le[256 + TNO - 1];
  __shared__ float li[256 + TNO - 1];
  __shared__ float eks[TNO];
  __shared__ float iks[TNO];
  const int s  = blockIdx.y;
  const int t0 = blockIdx.x * 256;
  const int tid = threadIdx.x;

  for (int j = tid; j < 256 + TNO - 1; j += 256) {
    int r = t0 - (TNO - 1) + j;
    bool ok = (r >= 0) && (r < TDATA);
    le[j] = ok ? aggE[(size_t)r * SUB + s] : 0.f;
    li[j] = ok ? aggI[(size_t)r * SUB + s] : 0.f;
  }
  for (int j = tid; j < TNO; j += 256) {
    eks[j] = EK[s * TNO + j];
    iks[j] = IK[s * TNO + j];
  }
  __syncthreads();

  int t = t0 + tid;
  if (t < TDATA) {
    float acc = Theta[s];
    #pragma unroll 4
    for (int tau = 0; tau < TNO; ++tau) {
      acc += le[tid + (TNO - 1) - tau] * eks[tau];
      acc += li[tid + (TNO - 1) - tau] * iks[tau];
    }
    SYN[(size_t)t * SUB + s] = acc;
  }
}

// ---------------------------------------------------------------------------
// Kernel 3: sequential GLM scan (single workgroup, 8 wave32 waves).
// Recurrent input accumulated by scatter into a circular LDS buffer with
// ds_add_f32 atomics; syn/U staged into LDS in chunks via the Tensor Data
// Mover (falls back to cooperative loads if the builtin is unavailable).
// ---------------------------------------------------------------------------
#define WIN 256           // circular future-window slots (>= TNO+1)
#define CHK 128           // syn/U staging chunk (time steps)
#define NT  16            // scatter tasks per thread (256*16 >= 20*200)

__global__ __launch_bounds__(256) void k_scan(
    const float* __restrict__ SYN, const float* __restrict__ U,
    const float* __restrict__ HK, const float* __restrict__ PK,
    const int* __restrict__ ccnt, const int* __restrict__ cidx,
    const float* __restrict__ cw, float* __restrict__ Z)
{
  __shared__ float fut[WIN * SUB];     // accumulated hist+prop per future time
  __shared__ float synb[CHK * SUB];    // staged syn (+Theta)
  __shared__ float ub[CHK * SUB];      // staged uniforms
  __shared__ unsigned smask;
  const int tid = threadIdx.x;

  // ---- preload per-task constants into registers ----
  float hval[NT]; int hcol[NT]; int hdly[NT];
  float pval[NT][4]; int pcs[NT][4]; int pcol[NT]; int pdly[NT];
  #pragma unroll
  for (int i = 0; i < NT; ++i) {
    int k = tid + 256 * i;
    bool v = (k < SUB * TNO);
    int c = v ? (k / TNO) : 0;
    int d = v ? (k % TNO) : 0;
    hval[i] = v ? HK[c * TNO + d] : 0.f;   // spike(c,t) -> fut[t+d+1][c]
    hcol[i] = c;
    hdly[i] = d + 1;
    pcol[i] = c;                            // target subunit for prop
    pdly[i] = d + 1;
    int cnt = v ? ccnt[c] : 0;
    #pragma unroll
    for (int j = 0; j < 4; ++j) {
      int cs = (j < cnt) ? cidx[c * 8 + j] : 0;
      pcs[i][j] = cs;
      pval[i][j] = (j < cnt) ? cw[c * 8 + j] * PK[cs * TNO + d] : 0.f;
    }
  }

  for (int idx = tid; idx < WIN * SUB; idx += 256) fut[idx] = 0.f;
#if HAVE_TDM
  if (tid < 32) {
    tdm_load_1d_f32(SYN, synb, CHK * SUB);
    tdm_load_1d_f32(U,   ub,   CHK * SUB);
    __builtin_amdgcn_s_wait_tensorcnt(0);
  }
#else
  for (int j = tid; j < CHK * SUB; j += 256) { synb[j] = SYN[j]; ub[j] = U[j]; }
#endif
  __syncthreads();

  for (int t = 0; t < TDATA; ++t) {
    // -------- phase 1: emit spikes (threads 0..19, wave 0) --------
    if (tid < SUB) {
      int slot = (t & (WIN - 1)) * SUB + tid;
      float acc = fut[slot];
      fut[slot] = 0.f;                       // recycle slot for time t+WIN
      int ci = (t & (CHK - 1)) * SUB + tid;
      float x = acc + synb[ci];              // syn + Theta + hist + prop
      float p = 1.f / (1.f + __expf(-x));
      float spk = (ub[ci] < p) ? 1.f : 0.f;
      Z[(size_t)t * SUB + tid] = spk;        // fire-and-forget store
      unsigned long long b = __ballot(spk > 0.f);
      if (tid == 0) smask = (unsigned)b;
    }
    __syncthreads();

    // -------- phase 2: scatter spike influence into the future --------
    unsigned m = smask;
    if (m) {
      #pragma unroll
      for (int i = 0; i < NT; ++i) {
        if ((m >> hcol[i]) & 1u)
          atomicAdd(&fut[((t + hdly[i]) & (WIN - 1)) * SUB + hcol[i]], hval[i]);
      }
      #pragma unroll
      for (int i = 0; i < NT; ++i) {
        float v = 0.f;
        #pragma unroll
        for (int j = 0; j < 4; ++j)
          v += ((m >> pcs[i][j]) & 1u) ? pval[i][j] : 0.f;
        if (v != 0.f)
          atomicAdd(&fut[((t + pdly[i]) & (WIN - 1)) * SUB + pcol[i]], v);
      }
    }
    __syncthreads();

    // -------- refill syn/U staging every CHK steps --------
    if ((t & (CHK - 1)) == (CHK - 1) && (t + 1) < TDATA) {
      int base = (t + 1) * SUB;
      int elems = TDATA * SUB - base;
      if (elems > CHK * SUB) elems = CHK * SUB;
#if HAVE_TDM
      if (tid < 32) {
        tdm_load_1d_f32(SYN + base, synb, elems);
        tdm_load_1d_f32(U + base,   ub,   elems);
        __builtin_amdgcn_s_wait_tensorcnt(0);
      }
#else
      for (int j = tid; j < elems; j += 256) {
        synb[j] = SYN[base + j];
        ub[j]   = U[base + j];
      }
      if (tid == 0 && base + 2 * CHK * SUB < TDATA * SUB)
        __builtin_prefetch(SYN + base + CHK * SUB, 0, 1);
#endif
      __syncthreads();
    }
  }
}

// ---------------------------------------------------------------------------
// Host-side launcher
// ---------------------------------------------------------------------------
extern "C" void kernel_launch(void* const* d_in, const int* in_sizes, int n_in,
                              void* d_out, int out_size, void* d_ws, size_t ws_size,
                              hipStream_t stream) {
  (void)in_sizes; (void)n_in; (void)out_size; (void)ws_size;
  const float* Se    = (const float*)d_in[0];   // (30000,1000)
  const float* Si    = (const float*)d_in[1];   // (30000,250)
  const float* U     = (const float*)d_in[2];   // (30000,20)
  const float* Wsyn  = (const float*)d_in[3];   // (20,13,2)
  const float* Whist = (const float*)d_in[4];   // (20,13)
  const float* Wprop = (const float*)d_in[5];   // (20,13)
  const float* Theta = (const float*)d_in[6];   // (20,)
  const float* Cden  = (const float*)d_in[7];   // (20,20)
  const float* Ce    = (const float*)d_in[8];   // (20,1000)
  const float* Ci    = (const float*)d_in[9];   // (20,250)
  float* Z = (float*)d_out;                     // (30000,20)

  float* wf = (float*)d_ws;
  float*     EK   = wf;                       // 4000
  float*     IK   = wf + 4000;                // 4000
  float*     HK   = wf + 8000;                // 4000
  float*     PK   = wf + 12000;               // 4000
  _Float16*  CeH  = (_Float16*)(wf + 16000);  // 32*1024 f16 (16384 floats)
  _Float16*  CiH  = (_Float16*)(wf + 32384);  // 32*256  f16 (4096 floats)
  float*     aggE = wf + 36480;               // 600000
  float*     aggI = aggE + 600000;            // 600000
  float*     SYN  = aggI + 600000;            // 600000
  float*     cw   = SYN + 600000;             // 160
  int*       wi   = (int*)(cw + 160);
  int*       ccnt = wi;                       // 32
  int*       cidx = wi + 32;                  // 160

  k_precompute<<<1, 256, 0, stream>>>(Wsyn, Whist, Wprop, Cden, Ce, Ci,
                                      EK, IK, HK, PK, CeH, CiH,
                                      ccnt, cidx, cw);
  k_agg_wmma<<<(TDATA / 16 + 7) / 8, 256, 0, stream>>>(Se, Si, CeH, CiH,
                                                       aggE, aggI);
  k_conv<<<dim3((TDATA + 255) / 256, SUB), 256, 0, stream>>>(aggE, aggI, EK, IK,
                                                             Theta, SYN);
  k_scan<<<1, 256, 0, stream>>>(SYN, U, HK, PK, ccnt, cidx, cw, Z);
}
